// GCNLinkPredictor_41901700939972
// MI455X (gfx1250) — matrix operand order repaired and checked by reading, not compiled, not run on hardware
//
#include <hip/hip_runtime.h>
#include <hip/hip_bf16.h>

// ---------------------------------------------------------------------------
// GCN link predictor for MI455X (gfx1250, wave32, WMMA).
//   1) row_ptr build (binary search over sorted adj_rows)
//   2) h0 = x@W1 + b1          (bf16 WMMA, f32 accum, bias fused)
//   3) h  = relu(spmm(h0))     (wave-per-node, coalesced gathers, no atomics)
//   4) z0 = h@W2 + b2          (bf16 WMMA)
//   5) z  = spmm(z0)
//   6) out = rowdot(z[src], z[dst])
// ---------------------------------------------------------------------------

typedef __attribute__((ext_vector_type(16))) __bf16 v16bf;
typedef __attribute__((ext_vector_type(8)))  __bf16 v8bf;
typedef __attribute__((ext_vector_type(8)))  float  v8f;
typedef __attribute__((ext_vector_type(4)))  float  v4f;
typedef __attribute__((ext_vector_type(2)))  float  v2f;

// float -> bf16 with round-to-nearest-even (bit-exact, no reliance on __bf16 casts)
__device__ __forceinline__ __bf16 f2bf(float f) {
    union { float f; unsigned u; } x; x.f = f;
    unsigned r = (x.u + 0x7FFFu + ((x.u >> 16) & 1u)) >> 16;
    union { unsigned short s; __bf16 b; } y; y.s = (unsigned short)r;
    return y.b;
}

// ---------------------------------------------------------------------------
// row_ptr[i] = lower_bound(adj_rows, i); i in [0, N]  (adj_rows sorted)
// ---------------------------------------------------------------------------
__global__ __launch_bounds__(256) void rowptr_kernel(const int* __restrict__ rows,
                                                     int* __restrict__ row_ptr,
                                                     int n, int nnz) {
    int i = blockIdx.x * blockDim.x + threadIdx.x;
    if (i > n) return;
    int lo = 0, hi = nnz;
    while (lo < hi) {
        int mid = (lo + hi) >> 1;
        if (rows[mid] < i) lo = mid + 1; else hi = mid;
    }
    row_ptr[i] = lo;
}

// ---------------------------------------------------------------------------
// Y[nrows,NOUT] = X[nrows,K] @ W[K,NOUT] + bias   (bf16 WMMA, f32 accum)
// One wave per 16-row stripe; W transposed+converted to bf16 in LDS.
// ---------------------------------------------------------------------------
template<int K, int NOUT>
__global__ __launch_bounds__(256) void gemm_bias_kernel(const float* __restrict__ X,
                                                        const float* __restrict__ W,
                                                        const float* __restrict__ bias,
                                                        float* __restrict__ Y,
                                                        int nrows) {
    __shared__ __bf16 sW[K * NOUT];   // [n][k], bf16: 64KB (256x128) / 16KB (128x64)

    // Cooperative load: W row-major [k][n] -> LDS transposed [n][k] as bf16
    for (int i = threadIdx.x; i < K * NOUT; i += blockDim.x) {
        int k = i / NOUT, n = i % NOUT;
        sW[n * K + k] = f2bf(W[i]);
    }
    __syncthreads();

    const int wid  = threadIdx.x >> 5;
    const int lane = threadIdx.x & 31;
    const int stripe   = blockIdx.x * (blockDim.x >> 5) + wid;
    const int nStripes = (nrows + 15) >> 4;
    if (stripe >= nStripes) return;

    const int r0   = stripe << 4;
    const int half = lane >> 4;     // 0 or 1
    const int m    = lane & 15;     // row-in-stripe (A), n-in-tile (B/C)

    const float* xrow = X + (long)(r0 + m) * K;

    constexpr int CT = NOUT / 16;   // column tiles
    constexpr int KS = K / 32;      // K steps
    v8f acc[CT] = {};

    for (int ks = 0; ks < KS; ++ks) {
        const int kb = ks * 32;
        // --- A fragment (16x32 bf16): element e -> K = (e/8)*16 + half*8 + e%8
        const int kA = kb + half * 8;
        v4f g0 = *(const v4f*)(xrow + kA);
        v4f g1 = *(const v4f*)(xrow + kA + 4);
        v4f g2 = *(const v4f*)(xrow + kA + 16);
        v4f g3 = *(const v4f*)(xrow + kA + 20);
        v16bf a;
        #pragma unroll
        for (int e = 0; e < 4; ++e) {
            a[e]      = f2bf(g0[e]);
            a[4 + e]  = f2bf(g1[e]);
            a[8 + e]  = f2bf(g2[e]);
            a[12 + e] = f2bf(g3[e]);
        }
        // --- B fragments from LDS: lane n = m, element e -> K = kb + (e/8)*16 + half*8 + e%8
        #pragma unroll
        for (int ct = 0; ct < CT; ++ct) {
            const int n = ct * 16 + m;
            const __bf16* bp = &sW[n * K + kb + half * 8];
            v8bf blo = *(const v8bf*)bp;
            v8bf bhi = *(const v8bf*)(bp + 16);
            v16bf b = __builtin_shufflevector(blo, bhi,
                        0,1,2,3,4,5,6,7,8,9,10,11,12,13,14,15);
            acc[ct] = __builtin_amdgcn_wmma_f32_16x16x32_bf16(
                          false, a, false, b, (short)0, acc[ct], false, false);
        }
    }

    // --- store with fused bias. C layout: elem r -> row r0 + r + 8*half, col = ct*16+m
    #pragma unroll
    for (int ct = 0; ct < CT; ++ct) {
        const int n = ct * 16 + m;
        const float bb = bias[n];
        #pragma unroll
        for (int r = 0; r < 8; ++r) {
            const int row = r0 + r + half * 8;
            Y[(long)row * NOUT + n] = acc[ct][r] + bb;
        }
    }
}

// ---------------------------------------------------------------------------
// out[i,:] = sum_e vals[e] * dense[cols[e],:] over e in [row_ptr[i], row_ptr[i+1])
// One wave per node. Edges staged 32 at a time; col/val wave-broadcast; the
// whole wave does a coalesced D*4-byte gather per edge. No atomics needed.
// ---------------------------------------------------------------------------
template<int V> struct VecT;
template<> struct VecT<4> { typedef v4f T; };
template<> struct VecT<2> { typedef v2f T; };

template<int D, bool RELU>
__global__ __launch_bounds__(256) void spmm_kernel(const int* __restrict__ row_ptr,
                                                   const int* __restrict__ cols,
                                                   const float* __restrict__ vals,
                                                   const float* __restrict__ dense,
                                                   float* __restrict__ out,
                                                   int nrows) {
    constexpr int V = D / 32;
    typedef typename VecT<V>::T VT;

    const int node = blockIdx.x * (blockDim.x >> 5) + (threadIdx.x >> 5);
    if (node >= nrows) return;
    const int lane = threadIdx.x & 31;

    const int start = row_ptr[node];
    const int end   = row_ptr[node + 1];

    VT acc = {};
    for (int e0 = start; e0 < end; e0 += 32) {
        const int cnt = min(32, end - e0);
        int c = 0; float v = 0.0f;
        if (lane < cnt) { c = cols[e0 + lane]; v = vals[e0 + lane]; }
        for (int j = 0; j < cnt; ++j) {
            const int   cj = __shfl(c, j);
            const float vj = __shfl(v, j);
            VT rv = *(const VT*)(dense + (long)cj * D + lane * V);
            acc += vj * rv;
        }
    }

    VT* op = (VT*)(out + (long)node * D + lane * V);
    if (RELU) {
        #pragma unroll
        for (int t = 0; t < V; ++t) acc[t] = fmaxf(acc[t], 0.0f);
    }
    *op = acc;
}

// ---------------------------------------------------------------------------
// out[q] = dot(z[src[q],:], z[dst[q],:]) over OUT=64 dims
// ---------------------------------------------------------------------------
__global__ __launch_bounds__(256) void decode_kernel(const int* __restrict__ ei,
                                                     const float* __restrict__ z,
                                                     float* __restrict__ out,
                                                     int eq) {
    int q = blockIdx.x * blockDim.x + threadIdx.x;
    if (q >= eq) return;
    const int s = ei[q];
    const int d = ei[eq + q];
    const v4f* zs = (const v4f*)(z + (long)s * 64);
    const v4f* zd = (const v4f*)(z + (long)d * 64);
    float sum = 0.0f;
    #pragma unroll
    for (int i = 0; i < 16; ++i) {
        v4f p = zs[i] * zd[i];
        sum += p[0] + p[1] + p[2] + p[3];
    }
    out[q] = sum;
}

// ---------------------------------------------------------------------------
extern "C" void kernel_launch(void* const* d_in, const int* in_sizes, int n_in,
                              void* d_out, int out_size, void* d_ws, size_t ws_size,
                              hipStream_t stream) {
    const float* x        = (const float*)d_in[0];
    const int*   adj_rows = (const int*)  d_in[1];
    const int*   adj_cols = (const int*)  d_in[2];
    const float* adj_vals = (const float*)d_in[3];
    const int*   edge_idx = (const int*)  d_in[4];
    const float* W1       = (const float*)d_in[5];
    const float* b1       = (const float*)d_in[6];
    const float* W2       = (const float*)d_in[7];
    const float* b2       = (const float*)d_in[8];
    float*       out      = (float*)d_out;

    constexpr int IN_F = 256, HID = 128, OUTD = 64;
    const int N   = in_sizes[0] / IN_F;
    const int NNZ = in_sizes[1];
    const int EQ  = in_sizes[4] / 2;

    // Workspace layout (f32 elements); h0/z0 and h/z share storage.
    float* h0 = (float*)d_ws;                   // N*HID
    float* h  = h0 + (size_t)N * HID;           // N*HID
    float* z0 = h0;                             // N*OUTD (reuses h0)
    float* z  = h;                              // N*OUTD (reuses h)
    int* row_ptr = (int*)(h + (size_t)N * HID); // N+1 ints

    const int stripes = (N + 15) >> 4;

    // 1) CSR row pointers from sorted COO rows
    rowptr_kernel<<<(N + 1 + 255) / 256, 256, 0, stream>>>(adj_rows, row_ptr, N, NNZ);

    // 2) h0 = x @ W1 + b1
    gemm_bias_kernel<IN_F, HID><<<(stripes + 7) / 8, 256, 0, stream>>>(x, W1, b1, h0, N);

    // 3) h = relu(spmm(h0))
    spmm_kernel<HID, true><<<(N + 7) / 8, 256, 0, stream>>>(row_ptr, adj_cols, adj_vals, h0, h, N);

    // 4) z0 = h @ W2 + b2
    gemm_bias_kernel<HID, OUTD><<<(stripes + 7) / 8, 256, 0, stream>>>(h, W2, b2, z0, N);

    // 5) z = spmm(z0)
    spmm_kernel<OUTD, false><<<(N + 7) / 8, 256, 0, stream>>>(row_ptr, adj_cols, adj_vals, z0, z, N);

    // 6) edge decode
    decode_kernel<<<(EQ + 255) / 256, 256, 0, stream>>>(edge_idx, z, out, EQ);
}
